// TemporalPatchAttention_22290880266620
// MI455X (gfx1250) — matrix operand order repaired and checked by reading, not compiled, not run on hardware
//
#include <hip/hip_runtime.h>

typedef __attribute__((ext_vector_type(16))) __bf16        v16bf;
typedef __attribute__((ext_vector_type(8)))  float         v8f;
typedef __attribute__((ext_vector_type(4)))  unsigned int  v4u;

#define WMMA_BF16(a, b, c) \
  __builtin_amdgcn_wmma_f32_16x16x32_bf16(false, (a), false, (b), (short)0, (c), false, false)

// Problem constants (from reference setup_inputs)
constexpr int CB     = 16;     // batch
constexpr int CC     = 256;    // channels (cin == cout == emb)
constexpr int CH     = 128, CW = 128, CHW = CH * CW;
constexpr int CT     = 64;     // tokens per 8x8 patch
constexpr int CM     = 128;    // performer feature dim m
constexpr int NPATCH = CB * 16 * 16;  // 4096
constexpr int LDX    = 264;    // padded LDS row stride (bf16) for proj X tile
constexpr int LDKP   = 72;     // padded row stride for kp^T [m][t]
constexpr int LDZ    = 136;    // padded row stride for Z^T  [c][m]

// A-fragment K offset for 16-bit A 16x32 (ISA 7.12.2)
__device__ __forceinline__ int kOffA(int v, int hi) {
  return ((v >> 2) << 4) + ((v & 3) << 1) + (hi << 3);
}
// B-fragment K offset for 16-bit B 32x16
__device__ __forceinline__ int kOffB(int v, int hi) {
  return ((v >> 2) << 4) + ((v & 3) << 1) + (hi << 3);
}
__device__ __forceinline__ unsigned int packbf(float lo, float hi2) {
  __bf16 a = (__bf16)lo, b = (__bf16)hi2;
  return (unsigned int)__builtin_bit_cast(unsigned short, a) |
         ((unsigned int)__builtin_bit_cast(unsigned short, b) << 16);
}
// Async DMA: global -> LDS, 16B per lane, tracked by ASYNCcnt
__device__ __forceinline__ void async_g2l_b128(unsigned lds_addr, const void* gaddr) {
  asm volatile("global_load_async_to_lds_b128 %0, %1, off"
               :: "v"(lds_addr), "v"(gaddr) : "memory");
}
__device__ __forceinline__ void wait_asynccnt0() {
  asm volatile("s_wait_asynccnt 0" ::: "memory");
}

// ---------------------------------------------------------------------------
// One-time weight conversion f32 -> bf16 (Wc | Wp | wmat), 163840 elements.
// grid = 640, block = 256
// ---------------------------------------------------------------------------
__global__ __launch_bounds__(256) void cvt_weights_kernel(
    const float* __restrict__ Wc, const float* __restrict__ Wp,
    const float* __restrict__ wm, __bf16* __restrict__ dst)
{
  const int idx = blockIdx.x * 256 + threadIdx.x;
  if (idx < 65536)       dst[idx] = (__bf16)Wc[idx];
  else if (idx < 131072) dst[idx] = (__bf16)Wp[idx];
  else if (idx < 163840) dst[idx] = (__bf16)wm[idx - 131072];
}

// ---------------------------------------------------------------------------
// Projection: out[o, pix] = sum_c W[o,c] * X[c, pix] + bias[o]
// One block = 16-pixel column x all 256 out-channels (X tile staged once in
// LDS -> single HBM pass).  8 waves x 2 o-tiles each.  Weights pre-converted
// to bf16 (L2-resident).  grid = (1024, 16), block = 256.
// ---------------------------------------------------------------------------
__global__ __launch_bounds__(256) void proj_pack_kernel(
    const float*  __restrict__ feat,  // [B][C][H*W] f32
    const __bf16* __restrict__ Wb,    // [C][C] bf16 (row = out-ch)
    const float*  __restrict__ bias,  // [C]
    __bf16* __restrict__ outp)        // [NPATCH][CT][CC] bf16
{
  __shared__ __bf16 sX[16 * LDX];     // [n=16][c=256] bf16, padded rows

  const int tid  = threadIdx.x;
  const int wv   = tid >> 5;
  const int lane = tid & 31;
  const int lm   = lane & 15;
  const int hi   = lane >> 4;
  const int b    = blockIdx.y;
  const int n0   = blockIdx.x << 4;   // 1024 pixel-tiles per batch

  const float* X = feat + (size_t)b * CC * CHW;

  // ---- stage X tile: coalesced NT loads (read-once), transpose into LDS
  {
    const int nl = tid & 15;          // pixel within tile
    const int cb = tid >> 4;          // channel base 0..15
    #pragma unroll
    for (int i = 0; i < 16; ++i) {
      const int c = cb + i * 16;
      const float v = __builtin_nontemporal_load(&X[(size_t)c * CHW + n0 + nl]);
      sX[nl * LDX + c] = (__bf16)v;
    }
  }
  __syncthreads();

  // pixel-dependent output addressing (same for both o-tiles of this wave)
  const int n     = n0 + lm;
  const int h     = n >> 7, w = n & 127;
  const int patch = (b * 16 + (h >> 3)) * 16 + (w >> 3);
  const int t     = ((h & 7) << 3) + (w & 7);

  #pragma unroll
  for (int ot = 0; ot < 2; ++ot) {
    const int o0 = (wv + ot * 8) << 4;          // 16 o-tiles over 8 waves
    const int o  = o0 + lm;                     // A row
    v8f acc = {};
    #pragma unroll
    for (int c0 = 0; c0 < CC; c0 += 32) {
      v16bf a, bb;
      #pragma unroll
      for (int v = 0; v < 8; ++v) {
        const int k = c0 + kOffA(v, hi);
        a[2 * v]     = Wb[o * CC + k];
        a[2 * v + 1] = Wb[o * CC + k + 1];
      }
      #pragma unroll
      for (int v = 0; v < 8; ++v) {
        const int k = c0 + kOffB(v, hi);
        bb[2 * v]     = sX[lm * LDX + k];
        bb[2 * v + 1] = sX[lm * LDX + k + 1];
      }
      acc = WMMA_BF16(a, bb, acc);
    }
    // epilogue: bias, pack 8 consecutive o-values, one 16B store
    const int obase = o0 + (hi << 3);
    v4u pu;
    #pragma unroll
    for (int r = 0; r < 4; ++r)
      pu[r] = packbf(acc[2 * r]     + bias[obase + 2 * r],
                     acc[2 * r + 1] + bias[obase + 2 * r + 1]);
    *(v4u*)(outp + ((size_t)patch * CT + t) * CC + obase) = pu;
  }
}

// ---------------------------------------------------------------------------
// Performer attention + layernorm, one workgroup per 8x8 patch.
// block = 256 (8 waves), grid = 4096
// ---------------------------------------------------------------------------
__global__ __launch_bounds__(256) void perf_attn_kernel(
    const __bf16* __restrict__ qws,   // [NPATCH][CT][CC]
    const __bf16* __restrict__ kws,   // [NPATCH][CT][CC]
    const __bf16* __restrict__ wmb,   // [CM][CC] bf16
    float* __restrict__ out)          // [B][C][H][W] f32
{
  __shared__ __align__(16) char smem[171264];
  __bf16* sQ   = (__bf16*)smem;                    // [64][256], dead after GEMM1
  __bf16* sK   = (__bf16*)(smem + 32768);          // [64][256], dead after GEMM2
  float*  sY   = (float*)smem;                     // [64][256] f32, aliases sQ+sK
  __bf16* sKPT = (__bf16*)(smem + 65536);          // [128][LDKP] phi(k)^T  (18432 B)
  __bf16* sQP  = (__bf16*)(smem + 83968);          // [64][128]  phi(q)     (16384 B)
  __bf16* sZT  = (__bf16*)(smem + 100352);         // [256][LDZ] kptv as [c][m] (69632 B)
  float*  sXDK = (float*)(smem + 169984);          // [64]
  float*  sXDQ = sXDK + 64;                        // [64]
  float*  sKS  = sXDQ + 64;                        // [128]
  float*  sD   = sKS + 128;                        // [64]

  const int tid  = threadIdx.x;
  const int wv   = tid >> 5, lane = tid & 31;
  const int lm   = lane & 15, hi = lane >> 4;
  const int patch = blockIdx.x;
  const size_t pbase = (size_t)patch * CT * CC;

  // ---- stage Q,K patch tiles into LDS via async DMA (ASYNCcnt path)
  {
    const unsigned lq0 = (unsigned)(uintptr_t)(void*)sQ + (unsigned)tid * 16u;
    const unsigned lk0 = (unsigned)(uintptr_t)(void*)sK + (unsigned)tid * 16u;
    const char* gq = (const char*)(qws + pbase) + (size_t)tid * 16;
    const char* gk = (const char*)(kws + pbase) + (size_t)tid * 16;
    #pragma unroll
    for (int i = 0; i < 8; ++i) {
      async_g2l_b128(lq0 + i * 4096u, gq + (size_t)i * 4096);
      async_g2l_b128(lk0 + i * 4096u, gk + (size_t)i * 4096);
    }
    wait_asynccnt0();
  }
  __syncthreads();

  // ---- xd = 0.5 * ||x_t||^2 per token (K: tid<64, Q: 64<=tid<128)
  if (tid < 128) {
    const int t = tid & 63;
    const __bf16* src = (tid < 64) ? (sK + t * CC) : (sQ + t * CC);
    float s = 0.f;
    for (int c = 0; c < CC; ++c) { float x = (float)src[c]; s += x * x; }
    if (tid < 64) sXDK[t] = 0.5f * s; else sXDQ[t] = 0.5f * s;
  }
  __syncthreads();

  // ---- GEMM1: wtx = X * w^T, then phi = exp(wtx - xd)/sqrt(m)
  // waves 0-3 -> K features (stored transposed [m][t]); waves 4-7 -> Q ([t][m])
  {
    const float rsqm = 0.08838834764831845f;  // 1/sqrt(128)
    const int g = wv & 3;
    const __bf16* src = (wv < 4) ? sK : sQ;
    const float*  xd  = (wv < 4) ? sXDK : sXDQ;
    for (int i = 0; i < 8; ++i) {
      const int tileId = i * 4 + g;              // 0..31
      const int tt = tileId >> 3, mt = tileId & 7;
      const int trow = tt * 16 + lm;             // token (A row)
      const int mcol = mt * 16 + lm;             // feature (B col)
      v8f acc = {};
      #pragma unroll
      for (int c0 = 0; c0 < CC; c0 += 32) {
        v16bf a, bb;
        #pragma unroll
        for (int v = 0; v < 8; ++v) {
          const int k = c0 + kOffA(v, hi);
          a[2 * v]     = src[trow * CC + k];
          a[2 * v + 1] = src[trow * CC + k + 1];
        }
        #pragma unroll
        for (int v = 0; v < 8; ++v) {
          const int k = c0 + kOffB(v, hi);       // B[k=c][n=m] = wmb[m][c]
          bb[2 * v]     = wmb[mcol * CC + k];
          bb[2 * v + 1] = wmb[mcol * CC + k + 1];
        }
        acc = WMMA_BF16(a, bb, acc);
      }
      const int tbase = tt * 16 + (hi << 3);
      if (wv < 4) {
        // kp stored transposed [m][t]: 8 consecutive t -> one packed 16B store
        v4u pu;
        #pragma unroll
        for (int r = 0; r < 4; ++r)
          pu[r] = packbf(__expf(acc[2*r]   - xd[tbase + 2*r])   * rsqm,
                         __expf(acc[2*r+1] - xd[tbase + 2*r+1]) * rsqm);
        *(v4u*)(sKPT + mcol * LDKP + tbase) = pu;
      } else {
        #pragma unroll
        for (int r = 0; r < 8; ++r)
          sQP[(tbase + r) * CM + mcol] =
              (__bf16)(__expf(acc[r] - xd[tbase + r]) * rsqm);
      }
    }
  }
  __syncthreads();

  // ---- ksum[m] = sum_t kp[t][m]  (contiguous rows of kp^T)
  if (tid < CM) {
    float s = 0.f;
    for (int t = 0; t < CT; ++t) s += (float)sKPT[tid * LDKP + t];
    sKS[tid] = s;
  }
  __syncthreads();
  // ---- D[t] = qp[t] . ksum
  if (tid < CT) {
    float s = 0.f;
    for (int m = 0; m < CM; ++m) s += (float)sQP[tid * CM + m] * sKS[m];
    sD[tid] = s;
  }
  __syncthreads();

  // ---- GEMM2: Z[m][c] = sum_t kp[t][m] * v[t][c]; stored as Z^T [c][m]
  for (int i = 0; i < 16; ++i) {
    const int tileId = i * 8 + wv;               // 0..127
    const int mt = tileId >> 4, ct = tileId & 15;
    const int mrow = mt * 16 + lm;
    const int ccol = ct * 16 + lm;
    v8f acc = {};
    #pragma unroll
    for (int t0 = 0; t0 < CT; t0 += 32) {
      v16bf a, bb;
      #pragma unroll
      for (int v = 0; v < 8; ++v) {              // A[m][t] = kp^T, contiguous pairs
        const int t = t0 + kOffA(v, hi);
        a[2 * v]     = sKPT[mrow * LDKP + t];
        a[2 * v + 1] = sKPT[mrow * LDKP + t + 1];
      }
      #pragma unroll
      for (int v = 0; v < 8; ++v) {              // B[t][c] = sK[t][c]
        const int t = t0 + kOffB(v, hi);
        bb[2 * v]     = sK[t * CC + ccol];
        bb[2 * v + 1] = sK[(t + 1) * CC + ccol];
      }
      acc = WMMA_BF16(a, bb, acc);
    }
    // 8 consecutive m for fixed c -> one packed 16B store into Z^T
    const int mbase = mt * 16 + (hi << 3);
    v4u pu;
    #pragma unroll
    for (int r = 0; r < 4; ++r)
      pu[r] = packbf(acc[2 * r], acc[2 * r + 1]);
    *(v4u*)(sZT + ccol * LDZ + mbase) = pu;
  }
  __syncthreads();

  // ---- GEMM3: y[t][c] = (sum_m qp[t][m] * Z[m][c]) / (D[t] + eps) -> sY
  for (int i = 0; i < 8; ++i) {
    const int tileId = i * 8 + wv;               // 0..63
    const int tt = tileId >> 4, ct = tileId & 15;
    const int trow = tt * 16 + lm;
    const int ccol = ct * 16 + lm;
    v8f acc = {};
    #pragma unroll
    for (int m0 = 0; m0 < CM; m0 += 32) {
      v16bf a, bb;
      #pragma unroll
      for (int v = 0; v < 8; ++v) {              // A[t][m] = sQP row-major
        const int k = m0 + kOffA(v, hi);
        a[2 * v]     = sQP[trow * CM + k];
        a[2 * v + 1] = sQP[trow * CM + k + 1];
      }
      #pragma unroll
      for (int v = 0; v < 8; ++v) {              // B[m][c] = Z^T[c][m], contiguous pairs
        const int m = m0 + kOffB(v, hi);
        bb[2 * v]     = sZT[ccol * LDZ + m];
        bb[2 * v + 1] = sZT[ccol * LDZ + m + 1];
      }
      acc = WMMA_BF16(a, bb, acc);
    }
    #pragma unroll
    for (int r = 0; r < 8; ++r) {
      const int t = tt * 16 + (hi << 3) + r;
      sY[t * CC + ccol] = acc[r] / (sD[t] + 1e-8f);
    }
  }
  __syncthreads();

  // ---- layernorm over the 64 tokens per channel + merged writeout (NT stores)
  {
    const int c = tid;                           // 256 threads = 256 channels
    float s = 0.f, ss = 0.f;
    #pragma unroll 8
    for (int t = 0; t < CT; ++t) {
      const float v = sY[t * CC + c];
      s += v; ss += v * v;
    }
    const float mu  = s * (1.f / 64.f);
    const float var = ss * (1.f / 64.f) - mu * mu;
    const float rs  = rsqrtf(var + 1e-5f);
    const int b  = patch >> 8;
    const int ph = (patch >> 4) & 15, pw = patch & 15;
    float* op = out + (((size_t)b * CC + c) * CH + ph * 8) * CW + pw * 8;
    for (int th = 0; th < 8; ++th) {
      #pragma unroll
      for (int tw = 0; tw < 8; ++tw)
        __builtin_nontemporal_store((sY[(th * 8 + tw) * CC + c] - mu) * rs, &op[tw]);
      op += CW;
    }
  }
}

// ---------------------------------------------------------------------------
extern "C" void kernel_launch(void* const* d_in, const int* in_sizes, int n_in,
                              void* d_out, int out_size, void* d_ws, size_t ws_size,
                              hipStream_t stream) {
  (void)in_sizes; (void)n_in; (void)out_size; (void)ws_size;
  const float* feat_curr = (const float*)d_in[0];
  const float* feat_prev = (const float*)d_in[1];
  const float* Wc        = (const float*)d_in[2];
  const float* bc        = (const float*)d_in[3];
  const float* Wp        = (const float*)d_in[4];
  const float* bp        = (const float*)d_in[5];
  const float* wmat      = (const float*)d_in[6];
  float* out = (float*)d_out;

  __bf16* qws = (__bf16*)d_ws;                               // 128 MB
  __bf16* kws = qws + (size_t)NPATCH * CT * CC;              // +128 MB
  __bf16* wcb = kws + (size_t)NPATCH * CT * CC;              // 65536 bf16
  __bf16* wpb = wcb + 65536;                                 // 65536 bf16
  __bf16* wmb = wpb + 65536;                                 // 32768 bf16

  cvt_weights_kernel<<<640, 256, 0, stream>>>(Wc, Wp, wmat, wcb);

  dim3 pgrid(1024, CB, 1);
  proj_pack_kernel<<<pgrid, 256, 0, stream>>>(feat_curr, wcb, bc, qws);
  proj_pack_kernel<<<pgrid, 256, 0, stream>>>(feat_prev, wpb, bp, kws);
  perf_attn_kernel<<<NPATCH, 256, 0, stream>>>(qws, kws, wmb, out);
}
